// ActQuantizerGS_1176821039704
// MI455X (gfx1250) — matrix operand order repaired and checked by reading, not compiled, not run on hardware
//
#include <hip/hip_runtime.h>

// CDNA5 / gfx1250, wave32. Group-wise Hadamard(128) rotation + symmetric int4
// quantization + nibble packing, with the Hadamard done on the WMMA path:
//   H[k,n] = (-1)^popcount(k&n)  -> B fragments synthesized in registers.

typedef __attribute__((ext_vector_type(16))) __bf16 v16bf;
typedef __attribute__((ext_vector_type(8)))  float  v8f;

#define GS 128
#define WAVES 4
#define LDS_STRIDE 132  // 128 + 4 floats of pad: 2-way-conflict-free h spill

__global__ __launch_bounds__(WAVES * 32) void hadq_int4_kernel(
    const float* __restrict__ x, float* __restrict__ out, int num_groups) {

  __shared__ float hbuf[WAVES * 16 * LDS_STRIDE];

  const int lane   = threadIdx.x & 31;
  const int wave   = threadIdx.x >> 5;
  const int nloc   = lane & 15;
  const int hiLane = lane >> 4;  // 0 for lanes 0-15, 1 for lanes 16-31

  const int  rowBase = (blockIdx.x * WAVES + wave) * 16;   // 16 groups per wave
  const bool active  = (rowBase + 16) <= num_groups;

  // ---- B (Hadamard) sign fragments, 32x16 bf16 tiles, built from popcount parity.
  // B k order used here (contiguous, same as A): lane n = lane&15 holds column n; K = e for
  // lanes 0-15, K = 16+e for lanes 16-31 (e = half-index 0..15 in v16bf).
  // Full H index: k = kloc + 32*qb, n = nloc + 16*t, so
  //   parity(k&n) = parity(e & nloc) ^ (kbit4 & t0) ^ parity(qb & t[2:1])
  // with kbit4 == hiLane. Only 4 distinct fragments are ever needed.
  v16bf Bf[4];
#pragma unroll
  for (int e = 0; e < 16; ++e) {
    const float s  = (__popc(e & nloc) & 1) ? -1.0f : 1.0f;
    const float s1 = hiLane ? -s : s;
    Bf[0][e] = (__bf16)s;      // t even, no global neg
    Bf[1][e] = (__bf16)s1;     // t odd
    Bf[2][e] = (__bf16)(-s);   // t even, global neg
    Bf[3][e] = (__bf16)(-s1);  // t odd,  global neg
  }

  v8f acc[8] = {};  // 8 column tiles of 16x16 f32

  if (active) {
    // A k order used here (contiguous, same as B): lane holds row m = lane&15;
    // K = hiLane*16 + e, plus 32*qb per K-block.
    const float* rowp = x + (size_t)(rowBase + nloc) * GS;
#pragma unroll
    for (int qb = 0; qb < 4; ++qb) {
      const int kb = qb * 32 + hiLane * 16;
      const float4 f0 = *(const float4*)(rowp + kb);
      const float4 f1 = *(const float4*)(rowp + kb + 4);
      const float4 f2 = *(const float4*)(rowp + kb + 8);
      const float4 f3 = *(const float4*)(rowp + kb + 12);
      v16bf a;
      a[0]  = (__bf16)f0.x; a[1]  = (__bf16)f0.y; a[2]  = (__bf16)f0.z; a[3]  = (__bf16)f0.w;
      a[4]  = (__bf16)f1.x; a[5]  = (__bf16)f1.y; a[6]  = (__bf16)f1.z; a[7]  = (__bf16)f1.w;
      a[8]  = (__bf16)f2.x; a[9]  = (__bf16)f2.y; a[10] = (__bf16)f2.z; a[11] = (__bf16)f2.w;
      a[12] = (__bf16)f3.x; a[13] = (__bf16)f3.y; a[14] = (__bf16)f3.z; a[15] = (__bf16)f3.w;

#pragma unroll
      for (int t = 0; t < 8; ++t) {
        const int sel = ((__popc(qb & (t >> 1)) & 1) << 1) | (t & 1);  // constant after unroll
        acc[t] = __builtin_amdgcn_wmma_f32_16x16x32_bf16(
            false, a, false, Bf[sel], (short)0, acc[t], false, false);
      }
    }

    // Spill h tile to LDS. D layout: lane holds col nloc, row r + 8*hiLane per VGPR r.
    float* wbuf = &hbuf[wave * 16 * LDS_STRIDE];
#pragma unroll
    for (int t = 0; t < 8; ++t) {
#pragma unroll
      for (int r = 0; r < 8; ++r)
        wbuf[(r + hiLane * 8) * LDS_STRIDE + nloc + t * 16] = acc[t][r];
    }
  }
  __syncthreads();

  // ---- Epilogue: lane pair (L, L+16) owns group row = rowBase + (lane&15);
  // lane L handles cols 0..63, lane L+16 handles cols 64..127.
  if (active) {
    const int row = rowBase + nloc;
    const float* wbuf =
        &hbuf[wave * 16 * LDS_STRIDE + nloc * LDS_STRIDE + hiLane * 64];

    float hv[64];
    float amax = 0.0f;
#pragma unroll
    for (int j = 0; j < 16; ++j) {
      const float4 v = *(const float4*)(wbuf + j * 4);
      hv[j * 4 + 0] = v.x; hv[j * 4 + 1] = v.y;
      hv[j * 4 + 2] = v.z; hv[j * 4 + 3] = v.w;
      amax = fmaxf(amax, fmaxf(fmaxf(fabsf(v.x), fabsf(v.y)),
                               fmaxf(fabsf(v.z), fabsf(v.w))));
    }
    amax = fmaxf(amax, __shfl_xor(amax, 16, 32));  // combine the two half-rows

    const float scale = (amax > 0.0f) ? (amax * (1.0f / 7.0f)) : 1.0f;
    const float inv   = (amax > 0.0f) ? (7.0f / amax) : 0.0f;

    // d_out as float*: [scales NG][zero_point NG][q 64*NG], flat concat.
    const size_t NG = (size_t)num_groups;
    float* qout = out + 2 * NG + (size_t)row * 64 + (size_t)hiLane * 32;
#pragma unroll
    for (int j = 0; j < 8; ++j) {
      float pb[4];
#pragma unroll
      for (int c = 0; c < 4; ++c) {
        const float q0 = fminf(7.0f, fmaxf(-8.0f, rintf(hv[j * 8 + 2 * c + 0] * inv)));
        const float q1 = fminf(7.0f, fmaxf(-8.0f, rintf(hv[j * 8 + 2 * c + 1] * inv)));
        const int byte = (((int)q0) & 0xF) | ((((int)q1) & 0xF) << 4);
        pb[c] = (float)byte;
      }
      *(float4*)(qout + j * 4) = make_float4(pb[0], pb[1], pb[2], pb[3]);
    }
    if (!hiLane) {
      out[row]      = scale * 0.08838834764831845f;  // fold in 1/sqrt(128)
      out[NG + row] = 0.0f;                          // zero_point
    }
  }
}

extern "C" void kernel_launch(void* const* d_in, const int* in_sizes, int n_in,
                              void* d_out, int out_size, void* d_ws, size_t ws_size,
                              hipStream_t stream) {
  (void)n_in; (void)out_size; (void)d_ws; (void)ws_size;
  const float* x  = (const float*)d_in[0];
  float*       out = (float*)d_out;
  const int n          = in_sizes[0];
  const int num_groups = n / GS;                 // 524288 for the reference shape
  const int gpb        = WAVES * 16;             // 64 groups per block
  const int blocks     = (num_groups + gpb - 1) / gpb;
  hadq_int4_kernel<<<blocks, WAVES * 32, 0, stream>>>(x, out, num_groups);
}
